// DistanceStructureDecoder_4063039062776
// MI455X (gfx1250) — matrix operand order repaired and checked by reading, not compiled, not run on hardware
//
#include <hip/hip_runtime.h>
#include <hip/hip_bf16.h>

typedef __attribute__((ext_vector_type(16))) _Float16 v16h;
typedef __attribute__((ext_vector_type(8)))  _Float16 v8h;
typedef __attribute__((ext_vector_type(8)))  float    v8f;

namespace {

constexpr int NNODE = 8192;
constexpr int KNBR  = 32;
constexpr int NPAIR = NNODE * KNBR;   // 262144
constexpr int NBLOCKS = NPAIR / 32;   // 8192 blocks, one wave, 32 pairs each
constexpr int KS1   = 15;             // K-steps of 32 for feature GEMM (480)

// workspace layout in halves (_Float16)
constexpr int OFF_WCAT = 0;           // 15*4 frags * 512 = 30720 halves
constexpr int OFF_W1   = 30720;       // 2*8 frags * 512 =  8192 halves
constexpr int OFF_W2   = 38912;       // 4*4 frags * 512 =  8192 halves
constexpr int WS_HALVES = 47104;

constexpr float RBF_STEP  = 22.0f / 15.0f;   // center spacing
constexpr float RBF_ISIG  = 16.0f / 22.0f;   // 1/sigma

union ABfrag { v16h v; v8h h[2]; };

struct __align__(16) TileSmem {
  _Float16 feat[32][480];   // 30720 B : concatenated feature tile (f16)
  float posi[32][16];       //  2048 B
  float posj[32][16];       //  2048 B
  float Ri[32][12];         //  1536 B : R[j*3+i] (columns are e1,e2,e3)
  float ti[32][4];          //   512 B
  float Rj[32][12];         //  1536 B
  float maskv[32];          //   128 B
  float dmapv[32];          //   128 B
  int   same[32];           //   128 B
  int   rel[32];            //   128 B
  _Float16 pairh[32][64];   //  4096 B : normalized pair tile (f16)
  _Float16 hbuf[32][128];   //  8192 B : hidden activations (f16)
};                          // 51200 B total

__device__ __forceinline__ void frame(const float* p, float* R, float* t) {
  float t0 = p[3], t1 = p[4], t2 = p[5];
  float v1x = p[6] - t0, v1y = p[7] - t1, v1z = p[8] - t2;
  float v2x = p[0] - t0, v2y = p[1] - t1, v2z = p[2] - t2;
  float n1 = sqrtf(v1x*v1x + v1y*v1y + v1z*v1z) + 1e-6f;
  float e1x = v1x/n1, e1y = v1y/n1, e1z = v1z/n1;
  float d = v2x*e1x + v2y*e1y + v2z*e1z;
  float u2x = v2x - d*e1x, u2y = v2y - d*e1y, u2z = v2z - d*e1z;
  float n2 = sqrtf(u2x*u2x + u2y*u2y + u2z*u2z) + 1e-6f;
  float e2x = u2x/n2, e2y = u2y/n2, e2z = u2z/n2;
  float e3x = e1y*e2z - e1z*e2y;
  float e3y = e1z*e2x - e1x*e2z;
  float e3z = e1x*e2y - e1y*e2x;
  R[0]=e1x; R[1]=e2x; R[2]=e3x;   // R[j*3+i], j = spatial row
  R[3]=e1y; R[4]=e2y; R[5]=e3y;
  R[6]=e1z; R[7]=e2z; R[8]=e3z;
  t[0]=t0; t[1]=t1; t[2]=t2;
}

__device__ __forceinline__ float gelu_tanh(float x) {
  float x3 = x*x*x;
  return 0.5f * x * (1.0f + tanhf(0.7978845608028654f * (x + 0.044715f * x3)));
}

// ---------------- weight fragment prep -----------------------------------
// B-fragment layout (ISA 7.12.2, 16-bit B 32x16): lane l holds column
// N = l&15; halves h<8 hold K = h + 8*(l>>4), h>=8 hold K = 16 + 8*(l>>4) + h-8. Stored contiguously:
// frag[(kstep*NT+nt)*32 + lane][16 halves].
__global__ void prep_weights(const float* __restrict__ W_dmap,
                             const float* __restrict__ W_dist,
                             const float* __restrict__ W_dir,
                             const float* __restrict__ W_vec,
                             const float* __restrict__ W_rot,
                             const float* __restrict__ W1,
                             const float* __restrict__ W2,
                             _Float16* __restrict__ ws) {
  int id = blockIdx.x * 256 + threadIdx.x;
  if (id >= WS_HALVES) return;
  if (id < OFF_W1) {                      // concatenated 480x64 weights
    int f = id >> 9, r = id & 511, lane = r >> 4, h = r & 15;
    int ks = f >> 2, nt = f & 3;
    int kk = ks*32 + (h & 7) + ((lane >> 4) << 3) + ((h >> 3) << 4);
    int col = nt*16 + (lane & 15);
    float v = 0.0f;
    if      (kk < 16)  v = W_dmap[kk*64 + col];
    else if (kk < 416) v = W_dist[(kk-16)*64 + col];
    else if (kk < 431) v = W_dir[(kk-416)*64 + col];
    else if (kk < 446) v = W_vec[(kk-431)*64 + col];
    else if (kk < 455) v = W_rot[(kk-446)*64 + col];
    ws[id] = (_Float16)v;
  } else if (id < OFF_W2) {               // W1: 64x128
    int e = id - OFF_W1;
    int f = e >> 9, r = e & 511, lane = r >> 4, h = r & 15;
    int ks = f >> 3, nt = f & 7;
    int kk = ks*32 + (h & 7) + ((lane >> 4) << 3) + ((h >> 3) << 4);
    int col = nt*16 + (lane & 15);
    ws[id] = (_Float16)W1[kk*128 + col];
  } else {                                // W2: 128x64
    int e = id - OFF_W2;
    int f = e >> 9, r = e & 511, lane = r >> 4, h = r & 15;
    int ks = f >> 2, nt = f & 3;
    int kk = ks*32 + (h & 7) + ((lane >> 4) << 3) + ((h >> 3) << 4);
    int col = nt*16 + (lane & 15);
    ws[id] = (_Float16)W2[kk*64 + col];
  }
}

// ---------------- main fused kernel (one wave, 32 pairs = 2 M-tiles) -----
__global__ void __launch_bounds__(32) pair_kernel(
    const float* __restrict__ pos, const float* __restrict__ dmap,
    const int* __restrict__ neighbours, const int* __restrict__ resi,
    const int* __restrict__ chain, const int* __restrict__ batch,
    const float* __restrict__ mask, const float* __restrict__ W_rel,
    const float* __restrict__ ln_g, const float* __restrict__ ln_b,
    const float* __restrict__ b1, const float* __restrict__ b2,
    const _Float16* __restrict__ frags, float* __restrict__ out) {
  __shared__ TileSmem s;
  const int lane = threadIdx.x;       // 0..31, single wave per block
  const int p0 = blockIdx.x * 32;
  const int g  = lane >> 4;           // lane group (C/D rows: m = v + 8*g)
  const int cl = lane & 15;           // column-within-ntile / A-row index

  // ---- stage 0: per-pair scalars, positions, frames (one pair/lane) ----
  {
    const int p = p0 + lane;
    const int n = p >> 5;
    const int k = p & 31;
    int nb = neighbours[n * KNBR + k];
    float valid = (nb != -1) ? 1.0f : 0.0f;
    int j = nb < 0 ? 0 : nb;
    float mv = mask[n] * mask[j] * valid;
    s.maskv[lane] = mv;
    s.same[lane]  = ((chain[j] == chain[n]) && (batch[j] == batch[n])) ? 1 : 0;
    int rr = resi[j] - resi[n];
    rr = rr < -32 ? -32 : (rr > 32 ? 32 : rr);
    s.rel[lane]   = rr + 32;
    s.dmapv[lane] = dmap[(size_t)n * NNODE + j];
    float pi[15], pj[15];
#pragma unroll
    for (int q = 0; q < 15; ++q) { pi[q] = pos[n*15+q]; pj[q] = pos[j*15+q]; }
#pragma unroll
    for (int q = 0; q < 15; ++q) { s.posi[lane][q] = pi[q]; s.posj[lane][q] = pj[q]; }
    float R[9], t[3];
    frame(pi, R, t);
#pragma unroll
    for (int q = 0; q < 9; ++q) s.Ri[lane][q] = R[q];
    s.ti[lane][0] = t[0]; s.ti[lane][1] = t[1]; s.ti[lane][2] = t[2];
    frame(pj, R, t);
#pragma unroll
    for (int q = 0; q < 9; ++q) s.Rj[lane][q] = R[q];
    out[(size_t)NPAIR * 64 + p] = mv;          // pair_mask output
  }
  __syncthreads();

  // ---- stage 1: build 32x480 f16 feature tile in LDS --------------------
  // cols: [0,16) dmap-RBF(masked) | [16,416) dist-RBF | [416,431) dirs
  //       [431,446) rel_vec/10    | [446,455) rot     | [455,480) zero pad
  for (int u = lane; u < 800; u += 32) {        // zero the pad
    s.feat[u / 25][455 + u % 25] = (_Float16)0.0f;
  }
  for (int u = lane; u < 512; u += 32) {        // dmap RBF
    int pr = u >> 4, b = u & 15;
    float z = (s.dmapv[pr] - (float)b * RBF_STEP) * RBF_ISIG;
    float val = (s.maskv[pr] > 0.0f) ? __expf(-z * z) : 0.0f;
    s.feat[pr][b] = (_Float16)val;
  }
  for (int u = lane; u < 800; u += 32) {        // atom-pair distance RBF
    int pr = u / 25, ap = u % 25, a = ap / 5, b = ap % 5;
    float dx = s.posi[pr][a*3+0] - s.posj[pr][b*3+0];
    float dy = s.posi[pr][a*3+1] - s.posj[pr][b*3+1];
    float dz = s.posi[pr][a*3+2] - s.posj[pr][b*3+2];
    float d = sqrtf(dx*dx + dy*dy + dz*dz);
    v8h r0, r1;
#pragma unroll
    for (int bin = 0; bin < 8; ++bin) {
      float z = (d - (float)bin * RBF_STEP) * RBF_ISIG;
      r0[bin] = (_Float16)__expf(-z * z);
    }
#pragma unroll
    for (int bin = 0; bin < 8; ++bin) {
      float z = (d - (float)(bin + 8) * RBF_STEP) * RBF_ISIG;
      r1[bin] = (_Float16)__expf(-z * z);
    }
    v8h* dst = (v8h*)&s.feat[pr][16 + ap * 16];
    dst[0] = r0; dst[1] = r1;
  }
  for (int u = lane; u < 160; u += 32) {        // dirs + rel_vec/10
    int pr = u / 5, a = u % 5;
    float vx = s.posj[pr][a*3+0] - s.ti[pr][0];
    float vy = s.posj[pr][a*3+1] - s.ti[pr][1];
    float vz = s.posj[pr][a*3+2] - s.ti[pr][2];
    float r0 = s.Ri[pr][0]*vx + s.Ri[pr][3]*vy + s.Ri[pr][6]*vz;
    float r1 = s.Ri[pr][1]*vx + s.Ri[pr][4]*vy + s.Ri[pr][7]*vz;
    float r2 = s.Ri[pr][2]*vx + s.Ri[pr][5]*vy + s.Ri[pr][8]*vz;
    float inv = 1.0f / (sqrtf(r0*r0 + r1*r1 + r2*r2) + 1e-6f);
    s.feat[pr][416 + a*3 + 0] = (_Float16)(r0 * inv);
    s.feat[pr][416 + a*3 + 1] = (_Float16)(r1 * inv);
    s.feat[pr][416 + a*3 + 2] = (_Float16)(r2 * inv);
    s.feat[pr][431 + a*3 + 0] = (_Float16)(r0 * 0.1f);
    s.feat[pr][431 + a*3 + 1] = (_Float16)(r1 * 0.1f);
    s.feat[pr][431 + a*3 + 2] = (_Float16)(r2 * 0.1f);
  }
  for (int u = lane; u < 288; u += 32) {        // rot = Ri^T * Rj (9 feats)
    int pr = u / 9, c = u % 9, i = c / 3, l = c % 3;
    float v = s.Ri[pr][0*3+i]*s.Rj[pr][0*3+l]
            + s.Ri[pr][1*3+i]*s.Rj[pr][1*3+l]
            + s.Ri[pr][2*3+i]*s.Rj[pr][2*3+l];
    s.feat[pr][446 + c] = (_Float16)v;
  }
  __syncthreads();

  // ---- GEMM1: pair(32x64) = feat(32x480) @ Wcat + W_rel one-hot seed ----
  // Two M-tiles (mt=0: rows 0-15, mt=1: rows 16-31) share each B-fragment.
  v8f acc[2][4];
#pragma unroll
  for (int mt = 0; mt < 2; ++mt) {
#pragma unroll
    for (int nt = 0; nt < 4; ++nt) {
      v8f c;
#pragma unroll
      for (int v = 0; v < 8; ++v) {
        int m = mt*16 + v + 8*g;
        c[v] = s.same[m] ? W_rel[(size_t)s.rel[m] * 64 + nt*16 + cl] : 0.0f;
      }
      acc[mt][nt] = c;
    }
  }
  for (int ks = 0; ks < KS1; ++ks) {
    ABfrag a0, a1;  // A-frag (16-bit 16x32): row = cl, halves per ISA layout
    const _Float16* ar0 = &s.feat[cl][0];
    const _Float16* ar1 = &s.feat[16 + cl][0];
    a0.h[0] = *(const v8h*)(ar0 + ks*32 + g*8);
    a0.h[1] = *(const v8h*)(ar0 + ks*32 + 16 + g*8);
    a1.h[0] = *(const v8h*)(ar1 + ks*32 + g*8);
    a1.h[1] = *(const v8h*)(ar1 + ks*32 + 16 + g*8);
#pragma unroll
    for (int nt = 0; nt < 4; ++nt) {
      ABfrag b;
      const _Float16* bp = frags + OFF_WCAT + ((size_t)(ks*4 + nt)*32 + lane)*16;
      b.h[0] = *(const v8h*)bp;
      b.h[1] = *(const v8h*)(bp + 8);
      acc[0][nt] = __builtin_amdgcn_wmma_f32_16x16x32_f16(
          false, a0.v, false, b.v, (short)0, acc[0][nt], false, false);
      acc[1][nt] = __builtin_amdgcn_wmma_f32_16x16x32_f16(
          false, a1.v, false, b.v, (short)0, acc[1][nt], false, false);
    }
  }

  // ---- LayerNorm over P=64 ---------------------------------------------
  float lg[4], lb[4];
#pragma unroll
  for (int nt = 0; nt < 4; ++nt) { lg[nt] = ln_g[nt*16 + cl]; lb[nt] = ln_b[nt*16 + cl]; }
#pragma unroll
  for (int mt = 0; mt < 2; ++mt) {
    v8f sm = acc[mt][0] + acc[mt][1] + acc[mt][2] + acc[mt][3];
    v8f sq = acc[mt][0]*acc[mt][0] + acc[mt][1]*acc[mt][1]
           + acc[mt][2]*acc[mt][2] + acc[mt][3]*acc[mt][3];
#pragma unroll
    for (int mb = 1; mb <= 8; mb <<= 1) {
#pragma unroll
      for (int v = 0; v < 8; ++v) {
        sm[v] += __shfl_xor(sm[v], mb, 32);
        sq[v] += __shfl_xor(sq[v], mb, 32);
      }
    }
#pragma unroll
    for (int v = 0; v < 8; ++v) {
      float mu  = sm[v] * (1.0f / 64.0f);
      float var = sq[v] * (1.0f / 64.0f) - mu * mu;
      float inv = rsqrtf(var + 1e-5f);
      int m = mt*16 + v + 8*g;
#pragma unroll
      for (int nt = 0; nt < 4; ++nt) {
        float y = lg[nt] * (acc[mt][nt][v] - mu) * inv + lb[nt];
        s.pairh[m][nt*16 + cl] = (_Float16)y;
      }
    }
  }
  __syncthreads();

  // ---- GEMM2: h(32x128) = pair(32x64) @ W1 + b1, GELU -------------------
  v8f hacc[2][8];
#pragma unroll
  for (int nt = 0; nt < 8; ++nt) {
    float bv = b1[nt*16 + cl];
    v8f c;
#pragma unroll
    for (int v = 0; v < 8; ++v) c[v] = bv;
    hacc[0][nt] = c;
    hacc[1][nt] = c;
  }
#pragma unroll
  for (int ks = 0; ks < 2; ++ks) {
    ABfrag a0, a1;
    const _Float16* ar0 = &s.pairh[cl][0];
    const _Float16* ar1 = &s.pairh[16 + cl][0];
    a0.h[0] = *(const v8h*)(ar0 + ks*32 + g*8);
    a0.h[1] = *(const v8h*)(ar0 + ks*32 + 16 + g*8);
    a1.h[0] = *(const v8h*)(ar1 + ks*32 + g*8);
    a1.h[1] = *(const v8h*)(ar1 + ks*32 + 16 + g*8);
#pragma unroll
    for (int nt = 0; nt < 8; ++nt) {
      ABfrag b;
      const _Float16* bp = frags + OFF_W1 + ((size_t)(ks*8 + nt)*32 + lane)*16;
      b.h[0] = *(const v8h*)bp;
      b.h[1] = *(const v8h*)(bp + 8);
      hacc[0][nt] = __builtin_amdgcn_wmma_f32_16x16x32_f16(
          false, a0.v, false, b.v, (short)0, hacc[0][nt], false, false);
      hacc[1][nt] = __builtin_amdgcn_wmma_f32_16x16x32_f16(
          false, a1.v, false, b.v, (short)0, hacc[1][nt], false, false);
    }
  }
#pragma unroll
  for (int mt = 0; mt < 2; ++mt) {
#pragma unroll
    for (int nt = 0; nt < 8; ++nt) {
#pragma unroll
      for (int v = 0; v < 8; ++v) {
        int m = mt*16 + v + 8*g;
        s.hbuf[m][nt*16 + cl] = (_Float16)gelu_tanh(hacc[mt][nt][v]);
      }
    }
  }
  __syncthreads();

  // ---- GEMM3: out(32x64) = h(32x128) @ W2 + b2 --------------------------
  v8f oacc[2][4];
#pragma unroll
  for (int nt = 0; nt < 4; ++nt) {
    float bv = b2[nt*16 + cl];
    v8f c;
#pragma unroll
    for (int v = 0; v < 8; ++v) c[v] = bv;
    oacc[0][nt] = c;
    oacc[1][nt] = c;
  }
#pragma unroll
  for (int ks = 0; ks < 4; ++ks) {
    ABfrag a0, a1;
    const _Float16* ar0 = &s.hbuf[cl][0];
    const _Float16* ar1 = &s.hbuf[16 + cl][0];
    a0.h[0] = *(const v8h*)(ar0 + ks*32 + g*8);
    a0.h[1] = *(const v8h*)(ar0 + ks*32 + 16 + g*8);
    a1.h[0] = *(const v8h*)(ar1 + ks*32 + g*8);
    a1.h[1] = *(const v8h*)(ar1 + ks*32 + 16 + g*8);
#pragma unroll
    for (int nt = 0; nt < 4; ++nt) {
      ABfrag b;
      const _Float16* bp = frags + OFF_W2 + ((size_t)(ks*4 + nt)*32 + lane)*16;
      b.h[0] = *(const v8h*)bp;
      b.h[1] = *(const v8h*)(bp + 8);
      oacc[0][nt] = __builtin_amdgcn_wmma_f32_16x16x32_f16(
          false, a0.v, false, b.v, (short)0, oacc[0][nt], false, false);
      oacc[1][nt] = __builtin_amdgcn_wmma_f32_16x16x32_f16(
          false, a1.v, false, b.v, (short)0, oacc[1][nt], false, false);
    }
  }
#pragma unroll
  for (int mt = 0; mt < 2; ++mt) {
#pragma unroll
    for (int nt = 0; nt < 4; ++nt) {
#pragma unroll
      for (int v = 0; v < 8; ++v) {
        int m = mt*16 + v + 8*g;
        out[(size_t)(p0 + m) * 64 + nt*16 + cl] = oacc[mt][nt][v];
      }
    }
  }
}

} // anonymous namespace

extern "C" void kernel_launch(void* const* d_in, const int* in_sizes, int n_in,
                              void* d_out, int out_size, void* d_ws, size_t ws_size,
                              hipStream_t stream) {
  (void)in_sizes; (void)n_in; (void)out_size; (void)ws_size;
  const float* pos        = (const float*)d_in[0];
  const float* dmap       = (const float*)d_in[1];
  const int*   neighbours = (const int*)d_in[2];
  const int*   resi       = (const int*)d_in[3];
  const int*   chain      = (const int*)d_in[4];
  const int*   batch      = (const int*)d_in[5];
  const float* mask       = (const float*)d_in[6];
  const float* W_rel      = (const float*)d_in[7];
  const float* W_dmap     = (const float*)d_in[8];
  const float* W_dist     = (const float*)d_in[9];
  const float* W_dir      = (const float*)d_in[10];
  const float* W_rot      = (const float*)d_in[11];
  const float* W_vec      = (const float*)d_in[12];
  const float* ln_g       = (const float*)d_in[13];
  const float* ln_b       = (const float*)d_in[14];
  const float* W1         = (const float*)d_in[15];
  const float* b1         = (const float*)d_in[16];
  const float* W2         = (const float*)d_in[17];
  const float* b2         = (const float*)d_in[18];
  float* out = (float*)d_out;
  _Float16* frags = (_Float16*)d_ws;

  prep_weights<<<(WS_HALVES + 255) / 256, 256, 0, stream>>>(
      W_dmap, W_dist, W_dir, W_vec, W_rot, W1, W2, frags);
  pair_kernel<<<NBLOCKS, 32, 0, stream>>>(
      pos, dmap, neighbours, resi, chain, batch, mask,
      W_rel, ln_g, ln_b, b1, b2, frags, out);
}
